// ChessCTM_54168127537256
// MI455X (gfx1250) — matrix operand order repaired and checked by reading, not compiled, not run on hardware
//
#include <hip/hip_runtime.h>
#include <hip/hip_bf16.h>

// ---------------- problem constants ----------------
static constexpr int B      = 128;
static constexpr int S      = 72;
static constexpr int ITER   = 16;
static constexpr int DMODEL = 1024;
static constexpr int DINPUT = 512;
static constexpr int MEMN   = 25;
static constexpr int HEADS  = 8;
static constexpr int DH     = 64;
static constexpr int NSO    = 256;
static constexpr int NSA    = 64;
static constexpr int NCLS   = 1968;
static constexpr int MH     = 16;
static constexpr int TED    = 128;
static constexpr int SA_SZ  = NSA * (NSA + 1) / 2;   // 2080
static constexpr int SO_SZ  = NSO * (NSO + 1) / 2;   // 32896
static constexpr int KSPLIT = 4;                     // 32896/32 = 1028 chunks = 4*257

typedef __attribute__((ext_vector_type(2)))  float  v2f;
typedef __attribute__((ext_vector_type(8)))  float  v8f;
typedef __attribute__((ext_vector_type(16))) __bf16 v16bf;
typedef __attribute__((ext_vector_type(4)))  int    v4i;

// -------- CDNA5 async global->LDS copy (guarded; fallback = sync copy) --------
#if __has_builtin(__builtin_amdgcn_global_load_async_to_lds_b128)
#define HAVE_ASYNC_LDS 1
typedef __attribute__((address_space(1))) v4i g1v4i;
typedef __attribute__((address_space(3))) v4i l3v4i;
#endif

__device__ __forceinline__ void async_cp16(const void* g, void* l) {
#ifdef HAVE_ASYNC_LDS
  __builtin_amdgcn_global_load_async_to_lds_b128((g1v4i*)g, (l3v4i*)l, 0, 0);
#else
  *(uint4*)l = *(const uint4*)g;
#endif
}

__device__ __forceinline__ void async_wait_all() {
#ifdef HAVE_ASYNC_LDS
#if __has_builtin(__builtin_amdgcn_s_wait_asynccnt)
  __builtin_amdgcn_s_wait_asynccnt(0);
#else
  asm volatile("s_wait_asynccnt 0x0" ::: "memory");
#endif
#endif
}

// ---------------- setup kernels ----------------

// triu index tables + decay->r + b recurrence init
__global__ void k_idx_setup(int N, int SZ, const float* __restrict__ decay,
                            int* __restrict__ ij, float* __restrict__ r,
                            float* __restrict__ b0, float binit) {
  int e = blockIdx.x * 256 + threadIdx.x;
  if (e >= SZ) return;
  float ef = (float)e;
  float tn = 2.0f * N + 1.0f;
  int i = (int)((tn - sqrtf(tn * tn - 8.0f * ef)) * 0.5f);
  if (i < 0) i = 0;
  if (i > N - 1) i = N - 1;
  auto base = [N](int ii) { return ii * (2 * N - ii + 1) / 2; };
  while (i > 0 && base(i) > e) --i;
  while (i < N - 1 && base(i + 1) <= e) ++i;
  int j = i + (e - base(i));
  ij[e] = (i << 16) | j;
  float d = fminf(fmaxf(decay[e], 0.0f), 15.0f);
  r[e] = expf(-d);
  b0[e] = binit;
}

__global__ void k_init_state(const float* __restrict__ start_act,
                             const float* __restrict__ start_trace,
                             float* __restrict__ act, float* __restrict__ trace) {
  int idx = blockIdx.x * 256 + threadIdx.x;
  if (idx >= B * DMODEL) return;
  int b = idx / DMODEL, d = idx % DMODEL;
  act[idx] = start_act[d];
  for (int m = 0; m < MEMN; m++)
    trace[((long)b * DMODEL + d) * MEMN + m] = start_trace[d * MEMN + m];
}

__global__ void k_init_ao(const float* __restrict__ start_act,
                          const int* __restrict__ ij_o,
                          float* __restrict__ a_o, float* __restrict__ a_a) {
  long idx = (long)blockIdx.x * 256 + threadIdx.x;
  if (idx < (long)B * SA_SZ) a_a[idx] = 0.0f;
  if (idx >= (long)B * SO_SZ) return;
  int e = (int)(idx % SO_SZ);
  int p = ij_o[e];
  a_o[idx] = start_act[p >> 16] * start_act[p & 0xffff];
}

// out_w (SO_SZ x NCLS f32) -> out_wT (NCLS x SO_SZ bf16), tiled transpose
__global__ void k_outw_cvt(const float* __restrict__ w, __hip_bfloat16* __restrict__ wt) {
  __shared__ float tile[16][17];
  int k0 = blockIdx.x * 16, n0 = blockIdx.y * 16;
  tile[threadIdx.y][threadIdx.x] = w[(long)(k0 + threadIdx.y) * NCLS + n0 + threadIdx.x];
  __syncthreads();
  wt[(long)(n0 + threadIdx.y) * SO_SZ + k0 + threadIdx.x] =
      __float2bfloat16(tile[threadIdx.x][threadIdx.y]);
}

// kv = LN(emb[x] @ kv_w + kv_b)  — one block per (b,s) row
__global__ void k_kv_embed(const int* __restrict__ x, const float* __restrict__ emb,
                           const float* __restrict__ kv_w, const float* __restrict__ kv_b,
                           const float* __restrict__ lng, const float* __restrict__ lnb,
                           float* __restrict__ kv) {
  __shared__ float e[TED];
  __shared__ float vals[DINPUT];
  __shared__ float rs[128], rq[128];
  int row = blockIdx.x, tid = threadIdx.x;
  int tok = x[row];
  if (tid < TED) e[tid] = emb[tok * TED + tid];
  __syncthreads();
  for (int d = tid; d < DINPUT; d += 128) {
    float acc = kv_b[d];
    for (int k = 0; k < TED; k++) acc += e[k] * kv_w[(long)k * DINPUT + d];
    vals[d] = acc;
  }
  __syncthreads();
  float s = 0.0f, s2 = 0.0f;
  for (int d = tid; d < DINPUT; d += 128) { float v = vals[d]; s += v; s2 += v * v; }
  rs[tid] = s; rq[tid] = s2; __syncthreads();
  for (int o = 64; o > 0; o >>= 1) {
    if (tid < o) { rs[tid] += rs[tid + o]; rq[tid] += rq[tid + o]; }
    __syncthreads();
  }
  float mean = rs[0] / DINPUT;
  float var  = rq[0] / DINPUT - mean * mean;
  float inv  = rsqrtf(var + 1e-5f);
  for (int d = tid; d < DINPUT; d += 128)
    kv[(long)row * DINPUT + d] = (vals[d] - mean) * inv * lng[d] + lnb[d];
}

// ---------------- generic fp32 WMMA GEMM ----------------
// C(MxN) = A(MxK,lda) @ B(KxN,ldb) + bias(N); block=256 (8 waves, each a 16x16 tile),
// grid = (N/16, M/128).  Full fp32 via V_WMMA_F32_16X16X4_F32.
__global__ void k_gemm_f32(const float* __restrict__ A, int lda,
                           const float* __restrict__ Bm, int ldb,
                           const float* __restrict__ bias,
                           float* __restrict__ C, int ldc, int K) {
  int lane = threadIdx.x & 31;
  int wave = threadIdx.x >> 5;
  int half = lane >> 4;
  int l15  = lane & 15;
  int m0   = blockIdx.y * 128 + wave * 16;
  int n    = blockIdx.x * 16 + l15;
  int koff = half * 2;
  const float* ap = A + (long)(m0 + l15) * lda + koff;
  const float* bp = Bm + (long)koff * ldb + n;
  const long bstep = (long)4 * ldb;
  v8f c = {};
  for (int k = 0; k < K; k += 4) {
    v2f a, b;
    a.x = ap[0];
    a.y = ap[1];
    b.x = bp[0];
    b.y = bp[ldb];
    c = __builtin_amdgcn_wmma_f32_16x16x4_f32(false, a, false, b, (short)0, c, false, false);
    ap += 4;
    bp += bstep;
  }
  float bv = bias[n];
#pragma unroll
  for (int r = 0; r < 8; r++)
    C[(long)(m0 + half * 8 + r) * ldc + n] = c[r] + bv;
}

// ---------------- attention (one block per (b,h)) ----------------
__global__ void k_attention(const float* __restrict__ qh, const float* __restrict__ KV,
                            float* __restrict__ o) {
  __shared__ float qs[DH];
  __shared__ float sc[S];
  __shared__ float inv_s;
  int bh = blockIdx.x, b = bh / HEADS, h = bh % HEADS;
  int tid = threadIdx.x;
  if (tid < DH) qs[tid] = qh[(long)b * DINPUT + h * DH + tid];
  __syncthreads();
  if (tid < S) {
    const float* Kr = KV + (long)(b * S + tid) * (2 * DINPUT) + h * DH;
    float acc = 0.0f;
    for (int d = 0; d < DH; d++) acc += qs[d] * Kr[d];
    sc[tid] = acc * 0.125f; // 1/sqrt(64)
  }
  __syncthreads();
  if (tid == 0) {
    float mx = sc[0];
    for (int s = 1; s < S; s++) mx = fmaxf(mx, sc[s]);
    float sm = 0.0f;
    for (int s = 0; s < S; s++) { float ev = expf(sc[s] - mx); sc[s] = ev; sm += ev; }
    inv_s = 1.0f / sm;
  }
  __syncthreads();
  if (tid < DH) {
    const float* Vb = KV + (long)(b * S) * (2 * DINPUT) + DINPUT + h * DH + tid;
    float acc = 0.0f;
    for (int s = 0; s < S; s++) acc += sc[s] * Vb[(long)s * (2 * DINPUT)];
    o[(long)b * DINPUT + h * DH + tid] = acc * inv_s;
  }
}

__global__ void k_concat_pre(const float* __restrict__ o2, const float* __restrict__ act,
                             float* __restrict__ pre) {
  int idx = blockIdx.x * 256 + threadIdx.x;
  if (idx >= B * (DINPUT + DMODEL)) return;
  int b = idx / (DINPUT + DMODEL), c = idx % (DINPUT + DMODEL);
  pre[idx] = (c < DINPUT) ? o2[(long)b * DINPUT + c]
                          : act[(long)b * DMODEL + (c - DINPUT)];
}

// state = LN(GLU(g)); write into circular trace slot
__global__ void k_glu_ln_trace(const float* __restrict__ g, const float* __restrict__ lng,
                               const float* __restrict__ lnb, float* __restrict__ trace,
                               int slot) {
  __shared__ float v[DMODEL];
  __shared__ float rs[256], rq[256];
  int b = blockIdx.x, tid = threadIdx.x;
  float s = 0.0f, s2 = 0.0f;
  for (int d = tid; d < DMODEL; d += 256) {
    float a    = g[(long)b * (2 * DMODEL) + d];
    float gate = g[(long)b * (2 * DMODEL) + DMODEL + d];
    float val  = a * (1.0f / (1.0f + expf(-gate)));
    v[d] = val; s += val; s2 += val * val;
  }
  rs[tid] = s; rq[tid] = s2; __syncthreads();
  for (int o = 128; o > 0; o >>= 1) {
    if (tid < o) { rs[tid] += rs[tid + o]; rq[tid] += rq[tid + o]; }
    __syncthreads();
  }
  float mean = rs[0] / DMODEL;
  float var  = rq[0] / DMODEL - mean * mean;
  float inv  = rsqrtf(var + 1e-5f);
  for (int d = tid; d < DMODEL; d += 256)
    trace[((long)b * DMODEL + d) * MEMN + slot] = (v[d] - mean) * inv * lng[d] + lnb[d];
}

// fused tp1 -> GLU -> tp2 -> GLU, per (b,d) thread, circular trace read
__global__ void k_trace_mlp(const float* __restrict__ trace, const float* __restrict__ tp1w,
                            const float* __restrict__ tp1b, const float* __restrict__ tp2w,
                            const float* __restrict__ tp2b, float* __restrict__ act,
                            int base) {
  int idx = blockIdx.x * 256 + threadIdx.x;
  if (idx >= B * DMODEL) return;
  int b = idx / DMODEL, d = idx % DMODEL;
  const float* tr = trace + ((long)b * DMODEL + d) * MEMN;
  float acc[2 * MH];
#pragma unroll
  for (int h = 0; h < 2 * MH; h++) acc[h] = tp1b[d * (2 * MH) + h];
  for (int m = 0; m < MEMN; m++) {
    int phys = base + m; if (phys >= MEMN) phys -= MEMN;
    float tv = tr[phys];
    const float* w = tp1w + ((long)m * (2 * MH)) * DMODEL + d;
#pragma unroll
    for (int h = 0; h < 2 * MH; h++) acc[h] += tv * w[(long)h * DMODEL];
  }
  float c0 = tp2b[d * 2 + 0], c1 = tp2b[d * 2 + 1];
#pragma unroll
  for (int h = 0; h < MH; h++) {
    float hv = acc[h] * (1.0f / (1.0f + expf(-acc[h + MH])));
    c0 += hv * tp2w[((long)h * 2 + 0) * DMODEL + d];
    c1 += hv * tp2w[((long)h * 2 + 1) * DMODEL + d];
  }
  act[idx] = c0 * (1.0f / (1.0f + expf(-c1)));
}

// ---------------- synchrony tensors ----------------
__global__ void k_sync_a(const float* __restrict__ act, const int* __restrict__ ij,
                         const float* __restrict__ r, const float* __restrict__ b_in,
                         float* __restrict__ b_out, float* __restrict__ a_a,
                         float* __restrict__ sync) {
  int idx = blockIdx.x * 256 + threadIdx.x;
  if (idx >= B * SA_SZ) return;
  int b = idx / SA_SZ, e = idx % SA_SZ;
  int p = ij[e];
  float si = act[(long)b * DMODEL + (DMODEL - NSA) + (p >> 16)];
  float sj = act[(long)b * DMODEL + (DMODEL - NSA) + (p & 0xffff)];
  float rv = r[e];
  float av = rv * a_a[idx] + si * sj;
  a_a[idx] = av;
  float bv = rv * b_in[e] + 1.0f;
  if (b == 0) b_out[e] = bv;
  sync[idx] = av / sqrtf(bv);
}

__global__ void k_sync_o(const float* __restrict__ act, const int* __restrict__ ij,
                         const float* __restrict__ r, const float* __restrict__ b_in,
                         float* __restrict__ b_out, float* __restrict__ a_o,
                         __hip_bfloat16* __restrict__ sync) {
  long idx = (long)blockIdx.x * 256 + threadIdx.x;
  if (idx >= (long)B * SO_SZ) return;
  int b = (int)(idx / SO_SZ), e = (int)(idx % SO_SZ);
  int p = ij[e];
  float si = act[(long)b * DMODEL + (p >> 16)];
  float sj = act[(long)b * DMODEL + (p & 0xffff)];
  float rv = r[e];
  float av = rv * a_o[idx] + si * sj;
  a_o[idx] = av;
  float bv = rv * b_in[e] + 1.0f;
  if (b == 0) b_out[e] = bv;
  sync[idx] = __float2bfloat16(av / sqrtf(bv));
}

// ---------------- big bf16 WMMA GEMM: split-K + LDS-staged B (async DMA) -------
// parts[s][m][n] partials of syncO(128 x 32896) @ out_wT^T (out_wT is NCLS x SO_SZ).
// Block tile 128x48 (8 waves x 16 rows, 3 n-tiles); B panel (48 cols x 32 k, 3KB)
// double-buffered in LDS via async global->LDS; each A fragment feeds 3 WMMAs.
__global__ __launch_bounds__(256) void k_gemm_big(const __hip_bfloat16* __restrict__ Abf,
                                                  const __hip_bfloat16* __restrict__ Bt,
                                                  float* __restrict__ parts) {
  constexpr int NT   = 3;                 // n-tiles per block
  constexpr int COLS = NT * 16;           // 48
  constexpr int CSTR = 40;                // bf16 per col in LDS (80B, 16B-aligned)
  __shared__ __hip_bfloat16 ldsB[2][COLS * CSTR];
  int tid  = threadIdx.x;
  int lane = tid & 31, wave = tid >> 5;
  int half = lane >> 4, l15 = lane & 15;
  int n0   = blockIdx.x * COLS;
  const int CHUNKS = SO_SZ / 32;          // 1028
  const int PER    = CHUNKS / KSPLIT;     // 257
  int cbase = blockIdx.y * PER;

  // staging map: tid<192 moves one 16B granule (col = tid/4, k-seg = tid%4)
  int scol = tid >> 2, sseg = tid & 3;
  const __hip_bfloat16* gB = Bt + (long)(n0 + scol) * SO_SZ + sseg * 8;
  __hip_bfloat16* lB0 = &ldsB[0][scol * CSTR + sseg * 8];
  __hip_bfloat16* lB1 = &ldsB[1][scol * CSTR + sseg * 8];

  const __hip_bfloat16* Ar = Abf + (long)(wave * 16 + l15) * SO_SZ;
  union Frag { uint4 q[2]; v16bf v; };
  v8f acc0 = {}, acc1 = {}, acc2 = {};

  if (tid < 192) async_cp16(gB + (long)cbase * 32, lB0);
  async_wait_all();
  __syncthreads();

  for (int c = 0; c < PER; c++) {
    int k0 = (cbase + c) * 32;
    const __hip_bfloat16* lbuf = ldsB[c & 1];
    if (c + 1 < PER && tid < 192)
      async_cp16(gB + (long)(cbase + c + 1) * 32, (c & 1) ? lB0 : lB1);
    Frag a;
    a.q[0] = *reinterpret_cast<const uint4*>(Ar + k0 + 16 * half);
    a.q[1] = *reinterpret_cast<const uint4*>(Ar + k0 + 16 * half + 8);
    Frag b0, b1, b2;
    b0.q[0] = *reinterpret_cast<const uint4*>(lbuf + (0 * 16 + l15) * CSTR + 16 * half);
    b0.q[1] = *reinterpret_cast<const uint4*>(lbuf + (0 * 16 + l15) * CSTR + 16 * half + 8);
    b1.q[0] = *reinterpret_cast<const uint4*>(lbuf + (1 * 16 + l15) * CSTR + 16 * half);
    b1.q[1] = *reinterpret_cast<const uint4*>(lbuf + (1 * 16 + l15) * CSTR + 16 * half + 8);
    b2.q[0] = *reinterpret_cast<const uint4*>(lbuf + (2 * 16 + l15) * CSTR + 16 * half);
    b2.q[1] = *reinterpret_cast<const uint4*>(lbuf + (2 * 16 + l15) * CSTR + 16 * half + 8);
    acc0 = __builtin_amdgcn_wmma_f32_16x16x32_bf16(false, a.v, false, b0.v, (short)0, acc0, false, false);
    acc1 = __builtin_amdgcn_wmma_f32_16x16x32_bf16(false, a.v, false, b1.v, (short)0, acc1, false, false);
    acc2 = __builtin_amdgcn_wmma_f32_16x16x32_bf16(false, a.v, false, b2.v, (short)0, acc2, false, false);
    async_wait_all();
    __syncthreads();
  }

  int mrow = wave * 16 + half * 8;
  float* P = parts + ((long)blockIdx.y * B + mrow) * NCLS + n0;
#pragma unroll
  for (int r = 0; r < 8; r++) {
    P[(long)r * NCLS + 0 * 16 + l15] = acc0[r];
    P[(long)r * NCLS + 1 * 16 + l15] = acc1[r];
    P[(long)r * NCLS + 2 * 16 + l15] = acc2[r];
  }
}

// reduce split-K partials + bias, write predictions (B,NCLS,ITER) and certainties (B,2,ITER)
__global__ void k_cert_write(const float* __restrict__ parts, const float* __restrict__ out_b,
                             float* __restrict__ dout, int t) {
  __shared__ float logits[NCLS];
  __shared__ float r1[256], r2[256];
  int b = blockIdx.x, tid = threadIdx.x;
  for (int n = tid; n < NCLS; n += 256) {
    float v = out_b[n];
#pragma unroll
    for (int s = 0; s < KSPLIT; s++) v += parts[((long)s * B + b) * NCLS + n];
    logits[n] = v;
    dout[((long)b * NCLS + n) * ITER + t] = v;
  }
  __syncthreads();
  float mx = -3.402823e38f;
  for (int n = tid; n < NCLS; n += 256) mx = fmaxf(mx, logits[n]);
  r1[tid] = mx; __syncthreads();
  for (int o = 128; o > 0; o >>= 1) {
    if (tid < o) r1[tid] = fmaxf(r1[tid], r1[tid + o]);
    __syncthreads();
  }
  mx = r1[0]; __syncthreads();
  float se = 0.0f, sxe = 0.0f;
  for (int n = tid; n < NCLS; n += 256) {
    float xm = logits[n] - mx;
    float ex = expf(xm);
    se += ex; sxe += ex * xm;
  }
  r1[tid] = se; r2[tid] = sxe; __syncthreads();
  for (int o = 128; o > 0; o >>= 1) {
    if (tid < o) { r1[tid] += r1[tid + o]; r2[tid] += r2[tid + o]; }
    __syncthreads();
  }
  if (tid == 0) {
    float L = logf(r1[0]);
    float mean_lp = r2[0] / r1[0] - L;             // sum p*lp
    float ne = -mean_lp / logf((float)NCLS);
    float* cert = dout + (long)B * NCLS * ITER;
    cert[(long)b * 2 * ITER + t]        = ne;
    cert[(long)b * 2 * ITER + ITER + t] = 1.0f - ne;
  }
}

// ---------------- host side ----------------
extern "C" void kernel_launch(void* const* d_in, const int* in_sizes, int n_in,
                              void* d_out, int out_size, void* d_ws, size_t ws_size,
                              hipStream_t stream) {
  (void)in_sizes; (void)n_in; (void)out_size; (void)ws_size;
  const int*   x          = (const int*)  d_in[0];
  const float* emb        = (const float*)d_in[1];
  const float* kv_w       = (const float*)d_in[2];
  const float* kv_b       = (const float*)d_in[3];
  const float* kv_ln_g    = (const float*)d_in[4];
  const float* kv_ln_b    = (const float*)d_in[5];
  const float* q_w        = (const float*)d_in[6];
  const float* q_b        = (const float*)d_in[7];
  const float* in_proj_w  = (const float*)d_in[8];
  const float* in_proj_b  = (const float*)d_in[9];
  const float* out_proj_w = (const float*)d_in[10];
  const float* out_proj_b = (const float*)d_in[11];
  const float* syn_w      = (const float*)d_in[12];
  const float* syn_b      = (const float*)d_in[13];
  const float* syn_ln_g   = (const float*)d_in[14];
  const float* syn_ln_b   = (const float*)d_in[15];
  const float* tp1_w      = (const float*)d_in[16];
  const float* tp1_b      = (const float*)d_in[17];
  const float* tp2_w      = (const float*)d_in[18];
  const float* tp2_b      = (const float*)d_in[19];
  const float* start_act  = (const float*)d_in[20];
  const float* start_tr   = (const float*)d_in[21];
  const float* decay_a    = (const float*)d_in[22];
  const float* decay_o    = (const float*)d_in[23];
  const float* out_w      = (const float*)d_in[24];
  const float* out_b      = (const float*)d_in[25];
  float* dout = (float*)d_out;

  // workspace carve-up
  char* base = (char*)d_ws;
  size_t off = 0;
  auto alloc = [&](size_t bytes) -> char* {
    char* p = base + off;
    off = (off + bytes + 255) & ~(size_t)255;
    return p;
  };
  __hip_bfloat16* out_wT = (__hip_bfloat16*)alloc((size_t)SO_SZ * NCLS * 2);
  float* KV      = (float*)alloc((size_t)B * S * 2 * DINPUT * 4);
  float* kv_tmp  = (float*)alloc((size_t)B * S * DINPUT * 4);
  float* trace   = (float*)alloc((size_t)B * DMODEL * MEMN * 4);
  float* a_o     = (float*)alloc((size_t)B * SO_SZ * 4);
  __hip_bfloat16* syncO = (__hip_bfloat16*)alloc((size_t)B * SO_SZ * 2);
  float* a_a     = (float*)alloc((size_t)B * SA_SZ * 4);
  float* syncA   = (float*)alloc((size_t)B * SA_SZ * 4);
  float* act     = (float*)alloc((size_t)B * DMODEL * 4);
  float* qbuf    = (float*)alloc((size_t)B * DINPUT * 4);
  float* qh      = (float*)alloc((size_t)B * DINPUT * 4);
  float* oat     = (float*)alloc((size_t)B * DINPUT * 4);
  float* o2      = (float*)alloc((size_t)B * DINPUT * 4);
  float* pre     = (float*)alloc((size_t)B * (DINPUT + DMODEL) * 4);
  float* g       = (float*)alloc((size_t)B * 2 * DMODEL * 4);
  float* parts   = (float*)alloc((size_t)KSPLIT * B * NCLS * 4);
  int*   ij_a    = (int*)  alloc((size_t)SA_SZ * 4);
  int*   ij_o    = (int*)  alloc((size_t)SO_SZ * 4);
  float* r_a     = (float*)alloc((size_t)SA_SZ * 4);
  float* r_o     = (float*)alloc((size_t)SO_SZ * 4);
  float* b_a0    = (float*)alloc((size_t)SA_SZ * 4);
  float* b_a1    = (float*)alloc((size_t)SA_SZ * 4);
  float* b_o0    = (float*)alloc((size_t)SO_SZ * 4);
  float* b_o1    = (float*)alloc((size_t)SO_SZ * 4);
  float* b_a_buf[2] = { b_a0, b_a1 };
  float* b_o_buf[2] = { b_o0, b_o1 };

  // ---- setup ----
  k_idx_setup<<<(SA_SZ + 255) / 256, 256, 0, stream>>>(NSA, SA_SZ, decay_a, ij_a, r_a, b_a0, 0.0f);
  k_idx_setup<<<(SO_SZ + 255) / 256, 256, 0, stream>>>(NSO, SO_SZ, decay_o, ij_o, r_o, b_o0, 1.0f);
  k_init_state<<<(B * DMODEL + 255) / 256, 256, 0, stream>>>(start_act, start_tr, act, trace);
  k_init_ao<<<(B * SO_SZ + 255) / 256, 256, 0, stream>>>(start_act, ij_o, a_o, a_a);
  k_outw_cvt<<<dim3(SO_SZ / 16, NCLS / 16), dim3(16, 16), 0, stream>>>(out_w, out_wT);
  k_kv_embed<<<B * S, 128, 0, stream>>>(x, emb, kv_w, kv_b, kv_ln_g, kv_ln_b, kv_tmp);
  // K,V = kv @ in_proj_w[:, 512:1536] + in_proj_b[512:1536]  (one GEMM, N=1024)
  k_gemm_f32<<<dim3((2 * DINPUT) / 16, (B * S) / 128), 256, 0, stream>>>(
      kv_tmp, DINPUT, in_proj_w + DINPUT, 3 * DINPUT, in_proj_b + DINPUT, KV, 2 * DINPUT, DINPUT);

  // ---- 16 recurrent iterations ----
  for (int t = 0; t < ITER; t++) {
    k_sync_a<<<(B * SA_SZ + 255) / 256, 256, 0, stream>>>(
        act, ij_a, r_a, b_a_buf[t & 1], b_a_buf[(t + 1) & 1], a_a, syncA);
    // q = sync_a @ q_w + q_b      (128 x 2080) @ (2080 x 512)
    k_gemm_f32<<<dim3(DINPUT / 16, 1), 256, 0, stream>>>(
        syncA, SA_SZ, q_w, DINPUT, q_b, qbuf, DINPUT, SA_SZ);
    // qh = q @ Wq + bq            (128 x 512) @ (512 x 512), Wq = in_proj_w[:, :512]
    k_gemm_f32<<<dim3(DINPUT / 16, 1), 256, 0, stream>>>(
        qbuf, DINPUT, in_proj_w, 3 * DINPUT, in_proj_b, qh, DINPUT, DINPUT);
    k_attention<<<B * HEADS, 128, 0, stream>>>(qh, KV, oat);
    // o2 = o @ out_proj_w + b
    k_gemm_f32<<<dim3(DINPUT / 16, 1), 256, 0, stream>>>(
        oat, DINPUT, out_proj_w, DINPUT, out_proj_b, o2, DINPUT, DINPUT);
    k_concat_pre<<<(B * (DINPUT + DMODEL) + 255) / 256, 256, 0, stream>>>(o2, act, pre);
    // g = pre @ syn_w + syn_b     (128 x 1536) @ (1536 x 2048)
    k_gemm_f32<<<dim3((2 * DMODEL) / 16, 1), 256, 0, stream>>>(
        pre, DINPUT + DMODEL, syn_w, 2 * DMODEL, syn_b, g, 2 * DMODEL, DINPUT + DMODEL);
    k_glu_ln_trace<<<B, 256, 0, stream>>>(g, syn_ln_g, syn_ln_b, trace, t % MEMN);
    k_trace_mlp<<<(B * DMODEL + 255) / 256, 256, 0, stream>>>(
        trace, tp1_w, tp1_b, tp2_w, tp2_b, act, (t + 1) % MEMN);
    k_sync_o<<<(B * SO_SZ + 255) / 256, 256, 0, stream>>>(
        act, ij_o, r_o, b_o_buf[t & 1], b_o_buf[(t + 1) & 1], a_o, syncO);
    // pred partials: (128 x 32896)bf16 @ (32896 x 1968)bf16, split-K=4, LDS-staged B
    k_gemm_big<<<dim3(NCLS / (3 * 16), KSPLIT), 256, 0, stream>>>(syncO, out_wT, parts);
    k_cert_write<<<B, 256, 0, stream>>>(parts, out_b, dout, t);
  }
}
